// ResidualEnsemble_22076131902008
// MI455X (gfx1250) — matrix operand order repaired and checked by reading, not compiled, not run on hardware
//
#include <hip/hip_runtime.h>

// ---------------- CDNA5 types ----------------
typedef __attribute__((ext_vector_type(16))) __bf16   v16bf;
typedef __attribute__((ext_vector_type(8)))  float    v8f;
typedef __attribute__((ext_vector_type(4)))  float    v4f;
typedef __attribute__((ext_vector_type(4)))  unsigned int v4u;

#define RVQ_DIM 256
#define RVQ_K   8192
#define GROUP_BYTES (16 * RVQ_DIM * 2)     // 16 codebook rows, bf16 = 8192 B
#define NGROUPS (RVQ_K / 16)               // 512
#define WG_THREADS 128                     // 4 waves
#define QUERIES_PER_WAVE 32                // two 16-row A tiles
#define QUERIES_PER_WG (4 * QUERIES_PER_WAVE)   // 128

__device__ __forceinline__ unsigned short f2bf(float x) {
    unsigned int u = __builtin_bit_cast(unsigned int, x);
    unsigned int r = u + 0x7FFFu + ((u >> 16) & 1u);   // round-to-nearest-even
    return (unsigned short)(r >> 16);
}

union BF16Frag {
    v16bf v;
    unsigned short s[16];
    v4u q[2];
};

// ---------------- fp32 -> bf16 codebook conversion (one pass) ----------------
__global__ __launch_bounds__(256) void cvt_bf16_kernel(const float* __restrict__ src,
                                                       unsigned int* __restrict__ dst,
                                                       long long n8) {
    long long i = (long long)blockIdx.x * blockDim.x + threadIdx.x;
    if (i >= n8) return;
    const v4f* sp = (const v4f*)src + i * 2;
    v4f f0 = sp[0];
    v4f f1 = sp[1];
    v4u o;
    o[0] = (unsigned)f2bf(f0[0]) | ((unsigned)f2bf(f0[1]) << 16);
    o[1] = (unsigned)f2bf(f0[2]) | ((unsigned)f2bf(f0[3]) << 16);
    o[2] = (unsigned)f2bf(f1[0]) | ((unsigned)f2bf(f1[1]) << 16);
    o[3] = (unsigned)f2bf(f1[2]) | ((unsigned)f2bf(f1[3]) << 16);
    ((v4u*)dst)[i] = o;
}

// ---------------- one RVQ stage ----------------------------------------------
// 4 waves x 32 queries = 128 queries per WG. B (16 codebook rows / group) is
// async-copied into a double-buffered LDS tile shared by all waves.
__global__ __launch_bounds__(WG_THREADS) void rvq_stage_kernel(
    float* __restrict__ res,                    // [N, DIM] fp32 residual (in/out)
    const unsigned short* __restrict__ cbb,     // [K, DIM] bf16 codebook (stage)
    const float* __restrict__ cbf,              // [K, DIM] fp32 codebook (stage)
    float* __restrict__ idx_out)                // [N] indices (as float)
{
    __shared__ unsigned char smem[2 * GROUP_BYTES];   // 16 KB double buffer

    const int tid   = threadIdx.x;
    const int wave  = tid >> 5;
    const int lane  = tid & 31;
    const int half  = lane >> 4;
    const int l16   = lane & 15;
    const int qbase = blockIdx.x * QUERIES_PER_WG + wave * QUERIES_PER_WAVE;

    // ---- Build A fragments for 2 tiles: 32x256 residual as bf16 -------------
    // A layout (16-bit 16x32): lane (l%16)=row M; half 0 covers K {0..7,16..23},
    // half 1 covers K {8..15,24..31}.
    v16bf a[2][8];
    #pragma unroll
    for (int t = 0; t < 2; ++t) {
        const float* rrow = res + (size_t)(qbase + t * 16 + l16) * RVQ_DIM;
        #pragma unroll
        for (int c = 0; c < 8; ++c) {
            BF16Frag f;
            const int b1 = c * 32 + half * 8;
            #pragma unroll
            for (int e = 0; e < 8; ++e) f.s[e]     = f2bf(rrow[b1 + e]);
            #pragma unroll
            for (int e = 0; e < 8; ++e) f.s[8 + e] = f2bf(rrow[b1 + 16 + e]);
            a[t][c] = f.v;
        }
    }

    float best[2][8];
    int   bidx[2][8];
    #pragma unroll
    for (int t = 0; t < 2; ++t)
        #pragma unroll
        for (int v = 0; v < 8; ++v) { best[t][v] = -3.402823466e38f; bidx[t][v] = 0; }

    const unsigned lds_base = (unsigned)(size_t)(&smem[0]);
    const unsigned long long gbase = (unsigned long long)(size_t)cbb;
    const int slice = tid * (GROUP_BYTES / WG_THREADS);   // 64 B per thread

    // Async-copy one 8KB group of 16 codebook rows into LDS buffer `buf`.
    auto issue_group = [&](int g, int buf) {
        unsigned lds = lds_base + (unsigned)buf * GROUP_BYTES + (unsigned)slice;
        unsigned long long ga = gbase + (unsigned long long)g * GROUP_BYTES
                                      + (unsigned long long)slice;
        asm volatile(
            "global_load_async_to_lds_b128 %0, %1, off\n\t"
            "global_load_async_to_lds_b128 %0, %1, off offset:16\n\t"
            "global_load_async_to_lds_b128 %0, %1, off offset:32\n\t"
            "global_load_async_to_lds_b128 %0, %1, off offset:48"
            :: "v"(lds), "v"(ga) : "memory");
    };

    issue_group(0, 0);
    asm volatile("s_wait_asynccnt 0x0" ::: "memory");
    __syncthreads();

    for (int g = 0; g < NGROUPS; ++g) {
        const int cur = g & 1;
        if (g + 1 < NGROUPS) issue_group(g + 1, cur ^ 1);   // prefetch next group

        // B layout (16-bit 32x16): lane = column N=(l%16); half 0 holds K {0..7,16..23},
        // half 1 holds K {8..15,24..31} (matches A).
        const unsigned short* bb = (const unsigned short*)(smem + cur * GROUP_BYTES)
                                   + (size_t)l16 * RVQ_DIM + half * 8;

        v8f acc0 = {};
        v8f acc1 = {};
        #pragma unroll
        for (int c = 0; c < 8; ++c) {
            BF16Frag b;
            b.q[0] = *(const v4u*)(bb + c * 32);       // ds_load_b128
            b.q[1] = *(const v4u*)(bb + c * 32 + 16);  // ds_load_b128
            acc0 = __builtin_amdgcn_wmma_f32_16x16x32_bf16(
                false, a[0][c], false, b.v, (short)0, acc0, false, false);
            acc1 = __builtin_amdgcn_wmma_f32_16x16x32_bf16(
                false, a[1][c], false, b.v, (short)0, acc1, false, false);
        }

        const int cand = g * 16 + l16;
        #pragma unroll
        for (int v = 0; v < 8; ++v) {
            if (acc0[v] > best[0][v]) { best[0][v] = acc0[v]; bidx[0][v] = cand; }
            if (acc1[v] > best[1][v]) { best[1][v] = acc1[v]; bidx[1][v] = cand; }
        }

        asm volatile("s_wait_asynccnt 0x0" ::: "memory");  // next-group data landed
        __syncthreads();                                   // all waves done with cur
    }

    // ---- Argmax across the 16 columns of each half (butterfly, low-idx tie) --
    #pragma unroll
    for (int m = 1; m < 16; m <<= 1) {
        #pragma unroll
        for (int t = 0; t < 2; ++t)
            #pragma unroll
            for (int v = 0; v < 8; ++v) {
                float ob = __shfl_xor(best[t][v], m, 16);
                int   oi = __shfl_xor(bidx[t][v], m, 16);
                if (ob > best[t][v] || (ob == best[t][v] && oi < bidx[t][v])) {
                    best[t][v] = ob; bidx[t][v] = oi;
                }
            }
    }
    // Every lane of half h now holds argmax for queries (tile t) qbase+t*16+h*8+v.

    if (l16 == 0) {
        #pragma unroll
        for (int t = 0; t < 2; ++t)
            #pragma unroll
            for (int v = 0; v < 8; ++v)
                idx_out[qbase + t * 16 + half * 8 + v] = (float)bidx[t][v];
    }

    // ---- Residual update in fp32: res[m] -= cbf[idx_m] ----------------------
    #pragma unroll
    for (int t = 0; t < 2; ++t) {
        #pragma unroll
        for (int m = 0; m < 16; ++m) {
            const int im = __shfl(bidx[t][m & 7], (m >> 3) * 16, 32);
            const v4f* crow = (const v4f*)(cbf + (size_t)im * RVQ_DIM);
            v4f* rr = (v4f*)(res + (size_t)(qbase + t * 16 + m) * RVQ_DIM);
            #pragma unroll
            for (int u = 0; u < 2; ++u) {
                const int e = lane + u * 32;        // 64 v4f per row / 32 lanes
                v4f rv = rr[e];
                v4f cv = crow[e];
                rv -= cv;
                rr[e] = rv;
            }
        }
    }
}

// ---------------- emb = query - final_residual ----------------
__global__ __launch_bounds__(256) void emb_kernel(const float* __restrict__ query,
                                                  const float* __restrict__ res,
                                                  float* __restrict__ emb,
                                                  long long n4) {
    long long i = (long long)blockIdx.x * blockDim.x + threadIdx.x;
    if (i >= n4) return;
    v4f q = ((const v4f*)query)[i];
    v4f r = ((const v4f*)res)[i];
    ((v4f*)emb)[i] = q - r;
}

extern "C" void kernel_launch(void* const* d_in, const int* in_sizes, int n_in,
                              void* d_out, int out_size, void* d_ws, size_t ws_size,
                              hipStream_t stream) {
    const float* query     = (const float*)d_in[0];     // [N, 256]
    const float* codebooks = (const float*)d_in[1];     // [D, 8192, 256]

    const int N = in_sizes[0] / RVQ_DIM;                 // 16384
    const int D = in_sizes[1] / (RVQ_K * RVQ_DIM);       // 4

    // Workspace: [res fp32 N*DIM] [codebooks bf16 D*K*DIM]
    float*          res = (float*)d_ws;
    unsigned short* cbb = (unsigned short*)((char*)d_ws + (size_t)N * RVQ_DIM * sizeof(float));

    // res <- query (d2d async: graph-capture safe)
    hipMemcpyAsync(res, query, (size_t)N * RVQ_DIM * sizeof(float),
                   hipMemcpyDeviceToDevice, stream);

    // Convert codebooks fp32 -> bf16 once (stays L2-resident)
    {
        long long n8 = (long long)D * RVQ_K * RVQ_DIM / 8;
        int blocks = (int)((n8 + 255) / 256);
        cvt_bf16_kernel<<<blocks, 256, 0, stream>>>(codebooks, (unsigned int*)cbb, n8);
    }

    // Output layout: [D*N] indices (as float), then [N*DIM] emb.
    float* idx_out = (float*)d_out;
    float* emb_out = (float*)d_out + (size_t)D * N;

    for (int d = 0; d < D; ++d) {
        rvq_stage_kernel<<<N / QUERIES_PER_WG, WG_THREADS, 0, stream>>>(
            res,
            cbb + (size_t)d * RVQ_K * RVQ_DIM,
            codebooks + (size_t)d * RVQ_K * RVQ_DIM,
            idx_out + (size_t)d * N);
    }

    // emb = query - residual
    {
        long long n4 = (long long)N * RVQ_DIM / 4;
        int blocks = (int)((n4 + 255) / 256);
        emb_kernel<<<blocks, 256, 0, stream>>>(query, res, emb_out, n4);
    }
}
